// MultiViewMamba_61083024884269
// MI455X (gfx1250) — matrix operand order repaired and checked by reading, not compiled, not run on hardware
//
#include <hip/hip_runtime.h>
#include <hip/hip_bf16.h>

// Problem constants (from reference): B=8, L=2048, V=4, D=512, N=16, DI=1024, K=4, R=32
#define BB 8
#define LL 2048
#define VV 4
#define DD 512
#define NN 16
#define DI 1024
#define KC 4
#define RR 32
#define MM (BB * LL) // 16384 rows per view

typedef __attribute__((ext_vector_type(16))) __bf16 v16bf;
typedef __attribute__((ext_vector_type(8)))  __bf16 v8bf;
typedef __attribute__((ext_vector_type(8)))  float  v8f;

__device__ __forceinline__ float sigmoidf_(float v) { return 1.f / (1.f + __expf(-v)); }

// ---------------------------------------------------------------------------
// Shared WMMA GEMM core: one 128(M) x NACC*16(N) block tile of
// C = A(MxK) * W(NxK)^T. 256 threads = 8 wave32; wave w owns rows
// [m0+16w, m0+16w+16) and all NACC 16-col accumulators
// (v_wmma_f32_16x16x32_bf16). A/B tiles staged through LDS.
// Fragment layouts follow CDNA5 ISA 7.12.2 (16-bit A 16x32; B 32x16 mirrored).
// NACC = 8 -> 128-wide N tile; NACC = 4 -> 64-wide.
// ---------------------------------------------------------------------------
template <int NACC>
__device__ __forceinline__ void gemm_tile(const __bf16* __restrict__ A,
                                          const __bf16* __restrict__ W,
                                          int lda, int ldb, int Ktot,
                                          int m0, int n0, v8f acc[NACC])
{
    __shared__ __bf16 As[128][40];        // 128x32 tile, 80B rows (16B aligned)
    __shared__ __bf16 Bs[NACC * 16][40];  // (NACC*16)x32 tile
    const int tid  = threadIdx.x;
    const int lane = tid & 31;
    const int wv   = tid >> 5;

    #pragma unroll
    for (int t = 0; t < NACC; ++t)
        #pragma unroll
        for (int i = 0; i < 8; ++i) acc[t][i] = 0.f;

    for (int k0 = 0; k0 < Ktot; k0 += 32) {
        // Stage A tile: 128x32 bf16, two 16B chunks per thread
        #pragma unroll
        for (int i = 0; i < 2; ++i) {
            int c = tid + 256 * i;
            int row = c >> 2, col = (c & 3) * 8;
            *(uint4*)&As[row][col] =
                *(const uint4*)&A[(size_t)(m0 + row) * lda + k0 + col];
        }
        // Stage B tile: (NACC*16)x32 bf16, NACC/4 16B chunks per thread
        #pragma unroll
        for (int i = 0; i < NACC / 4; ++i) {
            int c = tid + 256 * i;
            int row = c >> 2, col = (c & 3) * 8;
            *(uint4*)&Bs[row][col] =
                *(const uint4*)&W[(size_t)(n0 + row) * ldb + k0 + col];
        }
        // Warm L2 for the next k-tile (global_prefetch_b8)
        if (k0 + 32 < Ktot) {
            __builtin_prefetch(&A[(size_t)(m0 + (tid >> 2)) * lda + k0 + 32], 0, 0);
            __builtin_prefetch(&W[(size_t)(n0 + (tid >> 2) % (NACC * 16)) * ldb + k0 + 32], 0, 0);
        }
        __syncthreads();

        // A fragment: lane<16 -> K {0..7,16..23}; lane>=16 -> K {8..15,24..31}
        const int mr = wv * 16 + (lane & 15);
        const int ka = (lane < 16) ? 0 : 8;
        v8bf alo = *(const v8bf*)&As[mr][ka];
        v8bf ahi = *(const v8bf*)&As[mr][ka + 16];
        v16bf afrag;
        #pragma unroll
        for (int i = 0; i < 8; ++i) { afrag[i] = alo[i]; afrag[8 + i] = ahi[i]; }

        #pragma unroll
        for (int t = 0; t < NACC; ++t) {
            // B fragment: lane n holds column n; lane<16 -> K {0..7,16..23}; lane>=16 -> K {8..15,24..31}
            const int nr = t * 16 + (lane & 15);
            const int kb = (lane < 16) ? 0 : 8;
            v8bf blo = *(const v8bf*)&Bs[nr][kb];
            v8bf bhi = *(const v8bf*)&Bs[nr][kb + 16];
            v16bf bfrag;
            #pragma unroll
            for (int i = 0; i < 8; ++i) { bfrag[i] = blo[i]; bfrag[8 + i] = bhi[i]; }
            acc[t] = __builtin_amdgcn_wmma_f32_16x16x32_bf16(
                false, afrag, false, bfrag, (short)0, acc[t], false, false);
        }
        __syncthreads();
    }
}

// C/D layout (ISA 7.12.2): VGPR r -> M = r + (lane>=16 ? 8 : 0); N = lane%16.
#define EPILOGUE_COORDS(NT)                                                   \
    const int lane  = threadIdx.x & 31;                                       \
    const int wv    = threadIdx.x >> 5;                                       \
    const int mbase = blockIdx.x * 128 + wv * 16 + ((lane >= 16) ? 8 : 0);    \
    const int nbase = blockIdx.y * ((NT) * 16) + (lane & 15);

// ---------------------------------------------------------------------------
// Stage 0: f32 -> bf16 weight conversion
// ---------------------------------------------------------------------------
__global__ __launch_bounds__(256) void k_cvt(const float* __restrict__ s,
                                             __bf16* __restrict__ d, int n)
{
    int i = blockIdx.x * 256 + threadIdx.x;
    if (i < n) d[i] = (__bf16)s[i];
}

// ---------------------------------------------------------------------------
// Stage 1: LayerNorm over D per (b,l,v) row; one wave32 per row -> bf16 xn.
// xn layout: per-view contiguous [v][m][d], m = b*L+l.
// ---------------------------------------------------------------------------
__global__ __launch_bounds__(256) void k_ln(const float* __restrict__ x,
                                            const float* __restrict__ lw,
                                            const float* __restrict__ lb,
                                            __bf16* __restrict__ xn)
{
    const int wv = threadIdx.x >> 5, lane = threadIdx.x & 31;
    const int row = blockIdx.x * 8 + wv; // in [0, B*L*V)
    const int v = row % VV;
    const int bl = row / VV;
    const float* xr = x + (size_t)row * DD;

    float vals[DD / 32];
    float s = 0.f, s2 = 0.f;
    #pragma unroll
    for (int i = 0; i < DD / 32; ++i) {
        float t = xr[lane + i * 32];
        vals[i] = t; s += t; s2 += t * t;
    }
    #pragma unroll
    for (int off = 16; off >= 1; off >>= 1) {
        s  += __shfl_xor(s, off, 32);
        s2 += __shfl_xor(s2, off, 32);
    }
    const float mu   = s * (1.f / DD);
    const float var  = s2 * (1.f / DD) - mu * mu;
    const float rstd = rsqrtf(var + 1e-5f);

    __bf16* o = xn + (size_t)v * MM * DD + (size_t)bl * DD;
    const float* w = lw + v * DD;
    const float* b = lb + v * DD;
    #pragma unroll
    for (int i = 0; i < DD / 32; ++i) {
        int d = lane + i * 32;
        o[d] = (__bf16)((vals[i] - mu) * rstd * w[d] + b[d]);
    }
}

// ---------------------------------------------------------------------------
// Stage 2: xz = xn @ Win^T, split into xc (cols [0,DI)) and z (cols [DI,2DI))
// 128-wide N tiles (N=2048)
// ---------------------------------------------------------------------------
__global__ __launch_bounds__(256) void k_gemm_xz(const __bf16* __restrict__ xn,
                                                 const __bf16* __restrict__ wWin,
                                                 __bf16* __restrict__ xc,
                                                 __bf16* __restrict__ z)
{
    const int v = blockIdx.z;
    v8f acc[8];
    gemm_tile<8>(xn + (size_t)v * MM * DD, wWin + (size_t)v * 2 * DI * DD,
                 DD, DD, DD, blockIdx.x * 128, blockIdx.y * 128, acc);
    EPILOGUE_COORDS(8);
    __bf16* xcv_ = xc + (size_t)v * MM * DI;
    __bf16* zv   = z + (size_t)v * MM * DI;
    #pragma unroll
    for (int t = 0; t < 8; ++t) {
        const int ncol = nbase + t * 16;
        #pragma unroll
        for (int r = 0; r < 8; ++r) {
            const int m = mbase + r;
            const float val = acc[t][r];
            if (ncol < DI) xcv_[(size_t)m * DI + ncol] = (__bf16)val;
            else           zv[(size_t)m * DI + (ncol - DI)] = (__bf16)val;
        }
    }
}

// ---------------------------------------------------------------------------
// Stage 3: causal depthwise conv (K=4 taps) + bias + SiLU -> xcv
// ---------------------------------------------------------------------------
__global__ __launch_bounds__(256) void k_conv(const __bf16* __restrict__ xc,
                                              const float* __restrict__ Wc,
                                              const float* __restrict__ bc,
                                              __bf16* __restrict__ xcv)
{
    size_t g = (size_t)blockIdx.x * 256 + threadIdx.x;
    const int e = (int)(g % DI); g /= DI;
    const int l = (int)(g % LL); g /= LL;
    const int b = (int)(g % BB);
    const int v = (int)(g / BB);

    const __bf16* xv = xc + (size_t)v * MM * DI;
    float acc = bc[v * DI + e];
    #pragma unroll
    for (int k = 0; k < KC; ++k) {
        const int ls = l - (KC - 1) + k;
        if (ls >= 0)
            acc += Wc[(v * KC + k) * DI + e] *
                   (float)xv[((size_t)b * LL + ls) * DI + e];
    }
    xcv[(size_t)v * MM * DI + ((size_t)b * LL + l) * DI + e] =
        (__bf16)(acc * sigmoidf_(acc));
}

// ---------------------------------------------------------------------------
// Stage 4: dbl = xcv @ Wx^T  (N=64: dt|B|C) -> f32 (scan) + bf16 (next GEMM)
// 64-wide N tile (whole output width)
// ---------------------------------------------------------------------------
__global__ __launch_bounds__(256) void k_gemm_dbl(const __bf16* __restrict__ xcv,
                                                  const __bf16* __restrict__ wWx,
                                                  float* __restrict__ dblf,
                                                  __bf16* __restrict__ dblh)
{
    const int v = blockIdx.z;
    v8f acc[4];
    gemm_tile<4>(xcv + (size_t)v * MM * DI, wWx + (size_t)v * 64 * DI,
                 DI, DI, DI, blockIdx.x * 128, 0, acc);
    EPILOGUE_COORDS(4);
    float*  fo = dblf + (size_t)v * MM * 64;
    __bf16* ho = dblh + (size_t)v * MM * 64;
    #pragma unroll
    for (int t = 0; t < 4; ++t) {
        const int ncol = nbase + t * 16;
        #pragma unroll
        for (int r = 0; r < 8; ++r) {
            const int m = mbase + r;
            const float val = acc[t][r];
            fo[(size_t)m * 64 + ncol] = val;
            ho[(size_t)m * 64 + ncol] = (__bf16)val;
        }
    }
}

// ---------------------------------------------------------------------------
// Stage 5: delta = softplus(dt @ Wdt^T + bdt) -> bf16 (K=32, dt = dbl cols 0..31)
// 128-wide N tiles (N=1024)
// ---------------------------------------------------------------------------
__global__ __launch_bounds__(256) void k_gemm_delta(const __bf16* __restrict__ dblh,
                                                    const __bf16* __restrict__ wWdt,
                                                    const float* __restrict__ bdt,
                                                    __bf16* __restrict__ delta)
{
    const int v = blockIdx.z;
    v8f acc[8];
    gemm_tile<8>(dblh + (size_t)v * MM * 64, wWdt + (size_t)v * DI * RR,
                 64, RR, RR, blockIdx.x * 128, blockIdx.y * 128, acc);
    EPILOGUE_COORDS(8);
    __bf16* dv = delta + (size_t)v * MM * DI;
    #pragma unroll
    for (int t = 0; t < 8; ++t) {
        const int ncol = nbase + t * 16;
        const float bias = bdt[v * DI + ncol];
        #pragma unroll
        for (int r = 0; r < 8; ++r) {
            const int m = mbase + r;
            const float val = acc[t][r] + bias;
            const float sp = (val > 20.f) ? val : log1pf(__expf(val));
            dv[(size_t)m * DI + ncol] = (__bf16)sp;
        }
    }
}

// ---------------------------------------------------------------------------
// Stage 6: sequential SSM scan over L. One lane per channel e; 16 states in
// registers. B_t/C_t staged through LDS in chunks of 8 timesteps.
// Fuses D-skip and silu(z) gating -> bf16 y.
// ---------------------------------------------------------------------------
__global__ __launch_bounds__(256) void k_scan(const __bf16* __restrict__ delta,
                                              const __bf16* __restrict__ xcv,
                                              const float* __restrict__ dblf,
                                              const __bf16* __restrict__ z,
                                              const float* __restrict__ A_log,
                                              const float* __restrict__ Dp,
                                              __bf16* __restrict__ ybuf)
{
    const int c = blockIdx.x & 3;          // DI/256 = 4 blocks per (b,v)
    const int v = (blockIdx.x >> 2) & 3;
    const int b = blockIdx.x >> 4;
    const int e = c * 256 + threadIdx.x;

    float a[NN], h[NN];
    #pragma unroll
    for (int n = 0; n < NN; ++n) {
        a[n] = -__expf(A_log[((size_t)v * DI + e) * NN + n]);
        h[n] = 0.f;
    }
    const float dp = Dp[v * DI + e];

    const __bf16* dv  = delta + (size_t)v * MM * DI;
    const __bf16* xv  = xcv + (size_t)v * MM * DI;
    const __bf16* zv  = z + (size_t)v * MM * DI;
    const float*  dbv = dblf + (size_t)v * MM * 64;
    __bf16*       yv  = ybuf + (size_t)v * MM * DI;

    __shared__ float sB[8][NN], sC[8][NN];
    const int j = threadIdx.x >> 5, q = threadIdx.x & 31;

    for (int lc = 0; lc < LL; lc += 8) {
        {   // 256 threads = 8 timesteps x 32 (B|C) lanes
            const float val = dbv[((size_t)b * LL + lc + j) * 64 + RR + q];
            if (q < NN) sB[j][q] = val; else sC[j][q - NN] = val;
        }
        __syncthreads();
        for (int j2 = 0; j2 < 8; ++j2) {
            const size_t idx = ((size_t)b * LL + lc + j2) * DI + e;
            const float dlt  = (float)dv[idx];
            const float xval = (float)xv[idx];
            float y = 0.f;
            #pragma unroll
            for (int n = 0; n < NN; ++n) {
                const float dA = __expf(dlt * a[n]);
                h[n] = h[n] * dA + (dlt * sB[j2][n]) * xval;
                y += h[n] * sC[j2][n];
            }
            const float zz = (float)zv[idx];
            yv[idx] = (__bf16)((y + dp * xval) * (zz * sigmoidf_(zz)));
        }
        __syncthreads();
    }
}

// ---------------------------------------------------------------------------
// Stage 7: out = y @ Wout^T, written interleaved (B,L,V,D)
// 128-wide N tiles (N=512)
// ---------------------------------------------------------------------------
__global__ __launch_bounds__(256) void k_gemm_out(const __bf16* __restrict__ ybuf,
                                                  const __bf16* __restrict__ wWout,
                                                  float* __restrict__ out)
{
    const int v = blockIdx.z;
    v8f acc[8];
    gemm_tile<8>(ybuf + (size_t)v * MM * DI, wWout + (size_t)v * DD * DI,
                 DI, DI, DI, blockIdx.x * 128, blockIdx.y * 128, acc);
    EPILOGUE_COORDS(8);
    #pragma unroll
    for (int t = 0; t < 8; ++t) {
        const int ncol = nbase + t * 16;
        #pragma unroll
        for (int r = 0; r < 8; ++r) {
            const int m = mbase + r;
            out[((size_t)m * VV + v) * DD + ncol] = acc[t][r];
        }
    }
}

// ---------------------------------------------------------------------------
// Host launcher
// Inputs: 0:x 1:ln_w 2:ln_b 3:Win 4:Wconv 5:bconv 6:Wx 7:Wdt 8:bdt 9:A_log 10:Dp 11:Wout
// ---------------------------------------------------------------------------
extern "C" void kernel_launch(void* const* d_in, const int* in_sizes, int n_in,
                              void* d_out, int out_size, void* d_ws, size_t ws_size,
                              hipStream_t stream)
{
    const float* x     = (const float*)d_in[0];
    const float* ln_w  = (const float*)d_in[1];
    const float* ln_b  = (const float*)d_in[2];
    const float* Win   = (const float*)d_in[3];
    const float* Wconv = (const float*)d_in[4];
    const float* bconv = (const float*)d_in[5];
    const float* Wx    = (const float*)d_in[6];
    const float* Wdt   = (const float*)d_in[7];
    const float* bdt   = (const float*)d_in[8];
    const float* A_log = (const float*)d_in[9];
    const float* Dp    = (const float*)d_in[10];
    const float* Wout  = (const float*)d_in[11];
    float* out = (float*)d_out;

    char* p = (char*)d_ws;
    size_t off = 0;
    auto alloc = [&](size_t bytes) -> char* {
        char* r = p + off;
        off += (bytes + 255) & ~(size_t)255;
        return r;
    };
    __bf16* wWin  = (__bf16*)alloc((size_t)VV * 2 * DI * DD * 2);
    __bf16* wWx   = (__bf16*)alloc((size_t)VV * 64 * DI * 2);
    __bf16* wWdt  = (__bf16*)alloc((size_t)VV * DI * RR * 2);
    __bf16* wWout = (__bf16*)alloc((size_t)VV * DD * DI * 2);
    __bf16* xn    = (__bf16*)alloc((size_t)VV * MM * DD * 2);
    __bf16* xc    = (__bf16*)alloc((size_t)VV * MM * DI * 2);
    __bf16* zbuf  = (__bf16*)alloc((size_t)VV * MM * DI * 2);
    __bf16* xcv   = (__bf16*)alloc((size_t)VV * MM * DI * 2);
    float*  dblf  = (float*) alloc((size_t)VV * MM * 64 * 4);
    __bf16* dblh  = (__bf16*)alloc((size_t)VV * MM * 64 * 2);
    __bf16* delta = (__bf16*)alloc((size_t)VV * MM * DI * 2);
    __bf16* ybuf  = (__bf16*)alloc((size_t)VV * MM * DI * 2);

    // Stage 0: weight converts
    k_cvt<<<(VV * 2 * DI * DD + 255) / 256, 256, 0, stream>>>(Win, wWin, VV * 2 * DI * DD);
    k_cvt<<<(VV * 64 * DI + 255) / 256, 256, 0, stream>>>(Wx, wWx, VV * 64 * DI);
    k_cvt<<<(VV * DI * RR + 255) / 256, 256, 0, stream>>>(Wdt, wWdt, VV * DI * RR);
    k_cvt<<<(VV * DD * DI + 255) / 256, 256, 0, stream>>>(Wout, wWout, VV * DD * DI);

    // Stage 1: LayerNorm
    k_ln<<<(BB * LL * VV) / 8, 256, 0, stream>>>(x, ln_w, ln_b, xn);

    // Stage 2: xz GEMM (M=16384, N=2048, K=512)
    k_gemm_xz<<<dim3(MM / 128, (2 * DI) / 128, VV), 256, 0, stream>>>(xn, wWin, xc, zbuf);

    // Stage 3: conv + SiLU
    k_conv<<<(int)(((size_t)VV * MM * DI) / 256), 256, 0, stream>>>(xc, Wconv, bconv, xcv);

    // Stage 4: dbl GEMM (N=64, K=1024)
    k_gemm_dbl<<<dim3(MM / 128, 1, VV), 256, 0, stream>>>(xcv, wWx, dblf, dblh);

    // Stage 5: delta GEMM (N=1024, K=32) + softplus
    k_gemm_delta<<<dim3(MM / 128, DI / 128, VV), 256, 0, stream>>>(dblh, wWdt, bdt, delta);

    // Stage 6: SSM scan + gating
    k_scan<<<BB * VV * (DI / 256), 256, 0, stream>>>(delta, xcv, dblf, zbuf, A_log, Dp, ybuf);

    // Stage 7: output GEMM (N=512, K=1024)
    k_gemm_out<<<dim3(MM / 128, DD / 128, VV), 256, 0, stream>>>(ybuf, wWout, out);
}
